// MultiheadSelfAttention_89275190215037
// MI455X (gfx1250) — matrix operand order repaired and checked by reading, not compiled, not run on hardware
//
#include <hip/hip_runtime.h>
#include <hip/hip_bf16.h>

// ---------------------------------------------------------------------------
// MI455X (gfx1250) multihead-attention pipeline, f16 WMMA throughout.
//   B=4, T=1024, K=1024, H=16, S=64.  All matrix math uses
//   v_wmma_f32_16x16x32_f16 (f32 accumulate).
// ---------------------------------------------------------------------------

typedef _Float16 h16;
typedef __attribute__((ext_vector_type(8)))  _Float16 v8h;
typedef __attribute__((ext_vector_type(16))) _Float16 v16h;
typedef __attribute__((ext_vector_type(8)))  float    v8f;

#define WMMA_F16(a, b, c) \
  __builtin_amdgcn_wmma_f32_16x16x32_f16(false, (a), false, (b), (short)0, (c), false, false)

// --- A fragment (16x32, f16), row-major source, ld in elements. -------------
// ISA layout: lanes 0-15 = rows M, hi half-wave shifts K by 8.
//   elements 0..7  = src[k0 + hi*8 + 0..7]
//   elements 8..15 = src[k0 + 16 + hi*8 + 0..7]
// Two 16-byte loads per lane.
__device__ __forceinline__ v16h frag_a_load(const h16* __restrict__ base, int ld, int lane) {
  const int r = lane & 15, hi = lane >> 4;
  const h16* p = base + (size_t)r * ld + hi * 8;
  v8h lo = *(const v8h*)(p);
  v8h hf = *(const v8h*)(p + 16);
  return __builtin_shufflevector(lo, hf, 0,1,2,3,4,5,6,7,8,9,10,11,12,13,14,15);
}

// --- B fragment (32x16, f16) where B[k][n] = W[n0+n][k0+k], W row-major. ----
// ISA layout: lane = column N, hi half-wave shifts K by 8.
//   elements 0..7  = W[(n0+n)*ld + k0 + hi*8 + 0..7]
//   elements 8..15 = W[(n0+n)*ld + k0 + 16 + hi*8 + 0..7]
__device__ __forceinline__ v16h frag_bt_load(const h16* __restrict__ W, int n0, int k0,
                                             int ld, int lane) {
  const int n = lane & 15, hi = lane >> 4;
  const h16* p = W + (size_t)(n0 + n) * ld + k0 + hi * 8;
  v8h lo = *(const v8h*)(p);
  v8h hf = *(const v8h*)(p + 16);
  return __builtin_shufflevector(lo, hf, 0,1,2,3,4,5,6,7,8,9,10,11,12,13,14,15);
}

// ---------------------------------------------------------------------------
// Kernel 0: f32 -> f16 cast
// ---------------------------------------------------------------------------
__global__ __launch_bounds__(256) void cast_f32_f16(const float* __restrict__ src,
                                                    h16* __restrict__ dst, int n) {
  int i = blockIdx.x * 256 + threadIdx.x;
  if (i < n) dst[i] = (h16)src[i];
}

// ---------------------------------------------------------------------------
// Kernel 1: projection  O = x @ W^T, written head-split (B*H, T, 64) in f16.
// One wave computes a 16(M) x 64(N) tile; grid = (B*T/16, H).
// ---------------------------------------------------------------------------
__global__ __launch_bounds__(32) void proj_kernel(const h16* __restrict__ xh,
                                                  const h16* __restrict__ Wh,
                                                  h16* __restrict__ Oh) {
  const int lane = threadIdx.x & 31;
  const int r0   = blockIdx.x * 16;   // row within B*T = 4096
  const int head = blockIdx.y;        // 0..15 (column band of 64 = one head)

  v8f acc[4] = {};
  for (int k0 = 0; k0 < 1024; k0 += 32) {
    v16h a = frag_a_load(xh + (size_t)r0 * 1024 + k0, 1024, lane);
#pragma unroll
    for (int j = 0; j < 4; ++j) {
      v16h b = frag_bt_load(Wh, head * 64 + j * 16, k0, 1024, lane);
      acc[j] = WMMA_F16(a, b, acc[j]);
    }
  }
  // C layout: VGPR v -> M = v + 8*hi, N = lane&15.
  const int n = lane & 15, hi = lane >> 4;
#pragma unroll
  for (int j = 0; j < 4; ++j) {
#pragma unroll
    for (int v = 0; v < 8; ++v) {
      int m = v + hi * 8;
      int r = r0 + m;                 // global row over B*T
      int bidx = r >> 10, t = r & 1023;
      int s = j * 16 + n;             // element within head
      Oh[(((size_t)(bidx * 16 + head)) * 1024 + t) * 64 + s] = (h16)acc[j][v];
    }
  }
}

// ---------------------------------------------------------------------------
// Kernel 2: causal scores + softmax -> att (f16, (B*H, T, T)).
// One wave per (head, 16-row tile). 16 x 1024 f32 scores in LDS, rows padded
// to 1028 floats so same-part lanes land in distinct banks during the scans.
// Fully-masked causal column tiles are skipped entirely.
// ---------------------------------------------------------------------------
#define SC_LD 1028   // padded score row stride (floats)

__global__ __launch_bounds__(32) void attn_softmax_kernel(const h16* __restrict__ Qh,
                                                          const h16* __restrict__ Kh,
                                                          h16* __restrict__ att) {
  extern __shared__ float sc[];              // 16 * SC_LD f32
  __shared__ float red[32];
  __shared__ float rowmax[16], rowinv[16];

  const int lane = threadIdx.x & 31;
  const int t0   = blockIdx.x * 16;          // query rows
  const int head = blockIdx.y;               // 0..63 (B*H)
  const int n = lane & 15, hi = lane >> 4;

  const h16* qbase = Qh + ((size_t)head * 1024 + t0) * 64;
  const h16* kbase = Kh + (size_t)head * 1024 * 64;

  // Q fragments: 16 rows x 64 (two K-steps of 32)
  v16h a0 = frag_a_load(qbase + 0,  64, lane);
  v16h a1 = frag_a_load(qbase + 32, 64, lane);

  for (int u0 = 0; u0 <= t0 + 15; u0 += 16) {  // only tiles touching the causal triangle
    v16h b0 = frag_bt_load(kbase, u0, 0,  64, lane);  // B[s][u] = Kh[u][s]
    v16h b1 = frag_bt_load(kbase, u0, 32, 64, lane);
    v8f c = {};
    c = WMMA_F16(a0, b0, c);
    c = WMMA_F16(a1, b1, c);
#pragma unroll
    for (int v = 0; v < 8; ++v) {
      int m = v + hi * 8;
      int u = u0 + n;
      sc[m * SC_LD + u] = (u <= t0 + m) ? c[v] : -__builtin_huge_valf();
    }
  }
  __syncthreads();

  // Row-wise softmax over the valid prefix; two lanes per row.
  const int row = lane & 15, part = lane >> 4;
  const int vc = t0 + row + 1;               // valid count for this row
  float mx = -__builtin_huge_valf();
  for (int u = part; u < vc; u += 2) mx = fmaxf(mx, sc[row * SC_LD + u]);
  red[lane] = mx;
  __syncthreads();
  if (part == 0) rowmax[row] = fmaxf(red[row], red[row + 16]);
  __syncthreads();

  const float m = rowmax[row];
  float s = 0.0f;
  for (int u = part; u < vc; u += 2) {
    float e = __expf(sc[row * SC_LD + u] - m);
    sc[row * SC_LD + u] = e;                 // store numerator back
    s += e;
  }
  red[lane] = s;
  __syncthreads();
  if (part == 0) rowinv[row] = 1.0f / (red[row] + red[row + 16]);
  __syncthreads();

  // Coalesced normalized write; zeros above the diagonal.
  for (int mr = 0; mr < 16; ++mr) {
    const int t = t0 + mr;
    h16* orow = att + ((size_t)head * 1024 + t) * 1024;
    const float inv = rowinv[mr];
    for (int u = lane; u < 1024; u += 32)
      orow[u] = (u <= t) ? (h16)(sc[mr * SC_LD + u] * inv) : (h16)0.0f;
  }
}

// ---------------------------------------------------------------------------
// Kernel 3: out = att @ Wu^T + bu   (M=65536, N=1024, K=1024).
// Block = 4 waves; stages a 64x1024 f16 att tile in LDS via the CDNA5 async
// path (global_load_async_to_lds_b128 + s_wait_asynccnt), rows padded to
// 1032 halfs (2064 B) so frag_a_load's ds_load_b128 hits distinct banks.
// att is read from HBM exactly once; Wu streams from L2 as B^T fragments.
// ---------------------------------------------------------------------------
#define AT_LD 1032   // padded att-tile row stride (halfs); 2064 B, 16B aligned

__global__ __launch_bounds__(128) void att_unify_kernel(const h16* __restrict__ att,
                                                        const h16* __restrict__ Wuh,
                                                        const float* __restrict__ bu,
                                                        float* __restrict__ out) {
  extern __shared__ h16 s_att[];             // 64 * AT_LD halfs
  const int tid = threadIdx.x;
  const int lane = tid & 31, wave = tid >> 5;
  const size_t r0 = (size_t)blockIdx.x * 64; // rows over B*H*T

  // Stage 64 att rows (1024 halfs each) into LDS asynchronously.
  // 8192 b128 transfers; per-lane LDS byte offset + 64-bit global address.
  // Dynamic LDS is the only LDS in this kernel -> wave LDS offsets start at 0.
  {
    const char* gbase = (const char*)(att + r0 * 1024);
    for (int i = tid; i < 64 * 1024 / 8; i += 128) {     // 16B chunks
      int row = i >> 7, col = i & 127;                   // 128 chunks per row
      unsigned lds_off = (unsigned)(row * (AT_LD * 2) + col * 16);
      unsigned long long gaddr =
          (unsigned long long)(gbase + (size_t)row * 2048 + (size_t)col * 16);
      asm volatile("global_load_async_to_lds_b128 %0, %1, off"
                   :: "v"(lds_off), "v"(gaddr) : "memory");
    }
    asm volatile("s_wait_asynccnt 0" ::: "memory");
  }
  __syncthreads();

  const int mbase = wave * 16;               // this wave's 16 rows in the tile
  const int n = lane & 15, hi = lane >> 4;

  for (int nc = 0; nc < 16; ++nc) {          // 16 N-chunks of 64 columns
    const int n0 = nc * 64;
    v8f acc[4] = {};
    for (int k0 = 0; k0 < 1024; k0 += 32) {
      v16h a = frag_a_load(s_att + (size_t)mbase * AT_LD + k0, AT_LD, lane);
#pragma unroll
      for (int j = 0; j < 4; ++j) {
        v16h b = frag_bt_load(Wuh, n0 + j * 16, k0, 1024, lane);
        acc[j] = WMMA_F16(a, b, acc[j]);
      }
    }
#pragma unroll
    for (int j = 0; j < 4; ++j) {
#pragma unroll
      for (int v = 0; v < 8; ++v) {
        int m = v + hi * 8;
        int col = n0 + j * 16 + n;
        out[(r0 + mbase + m) * 1024 + col] = acc[j][v] + bu[col];
      }
    }
  }
}

// ---------------------------------------------------------------------------
// Host launch
// ---------------------------------------------------------------------------
extern "C" void kernel_launch(void* const* d_in, const int* in_sizes, int n_in,
                              void* d_out, int out_size, void* d_ws, size_t ws_size,
                              hipStream_t stream) {
  (void)in_sizes; (void)n_in; (void)out_size; (void)ws_size;

  const float* x  = (const float*)d_in[0];   // (4,1024,1024)
  const float* Wq = (const float*)d_in[1];   // (1024,1024)
  const float* Wk = (const float*)d_in[2];
  // d_in[3] = Wv: computed-but-unused in the reference -> skipped entirely.
  const float* Wu = (const float*)d_in[4];
  const float* bu = (const float*)d_in[5];   // (1024,)
  float* out = (float*)d_out;                // (64,1024,1024) f32

  // Workspace layout (f16 elements).
  const size_t N_X  = 4u * 1024 * 1024;      // x
  const size_t N_W  = 1024u * 1024;          // each weight
  const size_t N_QK = 64u * 1024 * 64;       // head-split Q / K
  h16* ws   = (h16*)d_ws;
  h16* xh   = ws;
  h16* Wqh  = xh  + N_X;
  h16* Wkh  = Wqh + N_W;
  h16* Wuh  = Wkh + N_W;
  h16* Qh   = Wuh + N_W;
  h16* Kh   = Qh  + N_QK;
  h16* atth = Kh  + N_QK;                    // 64*1024*1024 f16 = 128 MB

  // 0) casts
  cast_f32_f16<<<(int)((N_X + 255) / 256), 256, 0, stream>>>(x,  xh,  (int)N_X);
  cast_f32_f16<<<(int)((N_W + 255) / 256), 256, 0, stream>>>(Wq, Wqh, (int)N_W);
  cast_f32_f16<<<(int)((N_W + 255) / 256), 256, 0, stream>>>(Wk, Wkh, (int)N_W);
  cast_f32_f16<<<(int)((N_W + 255) / 256), 256, 0, stream>>>(Wu, Wuh, (int)N_W);

  // 1) Q/K projections (head-split outputs)
  {
    dim3 grid(4096 / 16, 16);                // (B*T)/16 row tiles x H column bands
    proj_kernel<<<grid, 32, 0, stream>>>(xh, Wqh, Qh);
    proj_kernel<<<grid, 32, 0, stream>>>(xh, Wkh, Kh);
  }

  // 2) causal scores + softmax -> att (f16)
  {
    dim3 grid(1024 / 16, 64);                // T/16 row tiles x (B*H) heads
    attn_softmax_kernel<<<grid, 32, 16 * SC_LD * sizeof(float), stream>>>(Qh, Kh, atth);
  }

  // 3) out = att @ Wu^T + bu
  {
    dim3 grid(65536 / 64);                   // M/64 row tiles
    att_unify_kernel<<<grid, 128, 64 * AT_LD * sizeof(h16), stream>>>(atth, Wuh, bu, out);
  }
}
